// WGCore_73701638799911
// MI455X (gfx1250) — matrix operand order repaired and checked by reading, not compiled, not run on hardware
//
#include <hip/hip_runtime.h>
#include <hip/hip_bf16.h>
#include <math.h>

// ---------------------------------------------------------------------------
// Types
// ---------------------------------------------------------------------------
typedef __attribute__((ext_vector_type(16))) __bf16       v16bf;
typedef __attribute__((ext_vector_type(8)))  float        v8f;
typedef __attribute__((ext_vector_type(4)))  float        v4f;
typedef __attribute__((ext_vector_type(4)))  unsigned int v4u;

#define MAX_ASEG 5

struct GemmParams {
  const float* a_ptr[MAX_ASEG];   // A segments: [M, a_len[s]] row-major, concatenated along K
  int          a_len[MAX_ASEG];
  int          n_a;
  const float* b_ptr[2];          // W segments: [N, b_len[s]] row-major (out = A @ W^T)
  int          b_len[2];
  int          n_b;
  const float* bias0;             // [N] or null
  const float* bias1;             // [N] or null
  float*       out;               // [M, N]
  int M, N, K;                    // all segment lengths are multiples of BK
};

union FragU {
  v16bf v;
  v4u   q[2];
};

// ---------------------------------------------------------------------------
// bf16 WMMA GEMM with segmented (fused-concat) A and W inputs.
// Block tile 128x128, K-chunk 32, 256 threads = 8 waves (4x2 wave grid),
// each wave computes a 32x64 tile = 8 x v_wmma_f32_16x16x32_bf16 per chunk.
// Segment resolution is hoisted out of the K loop (segment-window structure).
// ---------------------------------------------------------------------------
#define BM 128
#define BN 128
#define BK 32
#define LDP 40   // padded LDS row stride in bf16 elems (80B -> bank decorrelation)

__global__ __launch_bounds__(256) void gemm_bf16_wmma(GemmParams p) {
  __shared__ __bf16 As[BM][LDP];
  __shared__ __bf16 Bs[BN][LDP];

  const int tid    = threadIdx.x;
  const int lane   = tid & 31;
  const int wave   = tid >> 5;
  const int wm     = wave >> 1;                  // 0..3 (M direction, 32 rows each)
  const int wn     = wave & 1;                   // 0..1 (N direction, 64 cols each)
  const int blockM = blockIdx.y * BM;
  const int blockN = blockIdx.x * BN;

  // staging assignment: 256 threads x 16 elems for each 128x32 tile
  const int rT = tid >> 1;            // 0..127
  const int cT = (tid & 1) * 16;      // 0 or 16

  v8f acc[2][4] = {};

  int k = 0, a_st = 0, b_st = 0, sa = 0, sb = 0;
  while (k < p.K) {
    while (sa < p.n_a - 1 && k >= a_st + p.a_len[sa]) { a_st += p.a_len[sa]; ++sa; }
    while (sb < p.n_b - 1 && k >= b_st + p.b_len[sb]) { b_st += p.b_len[sb]; ++sb; }
    int wend = a_st + p.a_len[sa];
    {
      const int be = b_st + p.b_len[sb];
      if (be < wend) wend = be;
      if (p.K < wend) wend = p.K;
    }
    const float* aSrc = p.a_ptr[sa] + (size_t)(blockM + rT) * p.a_len[sa] + (k - a_st) + cT;
    const float* bSrc = p.b_ptr[sb] + (size_t)(blockN + rT) * p.b_len[sb] + (k - b_st) + cT;

    for (; k < wend; k += BK) {
      // ---- stage A and B 128x32 tiles, fp32 -> bf16 (native cvt)
      {
        const v4f* s4 = (const v4f*)aSrc;
        #pragma unroll
        for (int j = 0; j < 4; ++j) {
          v4f d = s4[j];
          As[rT][cT + 4 * j + 0] = (__bf16)d[0];
          As[rT][cT + 4 * j + 1] = (__bf16)d[1];
          As[rT][cT + 4 * j + 2] = (__bf16)d[2];
          As[rT][cT + 4 * j + 3] = (__bf16)d[3];
        }
        s4 = (const v4f*)bSrc;
        #pragma unroll
        for (int j = 0; j < 4; ++j) {
          v4f d = s4[j];
          Bs[rT][cT + 4 * j + 0] = (__bf16)d[0];
          Bs[rT][cT + 4 * j + 1] = (__bf16)d[1];
          Bs[rT][cT + 4 * j + 2] = (__bf16)d[2];
          Bs[rT][cT + 4 * j + 3] = (__bf16)d[3];
        }
      }
      aSrc += BK;
      bSrc += BK;
      __syncthreads();

      // ---- fragment loads per ISA 16-bit layouts
      FragU fa[2], fb[4];
      const int arow  = lane & 15;
      const int koffA = (lane >> 4) * 8;   // A: lanes<16 -> K{0..7,16..23}; hi -> K{8..15,24..31}
      #pragma unroll
      for (int ma = 0; ma < 2; ++ma) {
        const int r = wm * 32 + ma * 16 + arow;
        fa[ma].q[0] = *(const v4u*)&As[r][koffA];
        fa[ma].q[1] = *(const v4u*)&As[r][koffA + 16];
      }
      const int koffB = (lane >> 4) * 8;   // B: lanes<16 -> K{0..7,16..23}; hi -> K{8..15,24..31}
      #pragma unroll
      for (int nb = 0; nb < 4; ++nb) {
        const int r = wn * 64 + nb * 16 + (lane & 15);
        fb[nb].q[0] = *(const v4u*)&Bs[r][koffB];
        fb[nb].q[1] = *(const v4u*)&Bs[r][koffB + 16];
      }

      #pragma unroll
      for (int ma = 0; ma < 2; ++ma)
        #pragma unroll
        for (int nb = 0; nb < 4; ++nb)
          acc[ma][nb] = __builtin_amdgcn_wmma_f32_16x16x32_bf16(
              false, fa[ma].v, false, fb[nb].v, (short)0, acc[ma][nb], false, false);

      __syncthreads();
    }
  }

  // ---- epilogue: D layout -> lane n = lane&15, VGPR v -> M = v (+8 for hi lanes)
  const int mbase = blockM + wm * 32 + ((lane >> 4) << 3);
  const int nlane = lane & 15;
  #pragma unroll
  for (int nb = 0; nb < 4; ++nb) {
    const int n = blockN + wn * 64 + nb * 16 + nlane;
    float bsum = 0.f;
    if (p.bias0) bsum += p.bias0[n];
    if (p.bias1) bsum += p.bias1[n];
    #pragma unroll
    for (int ma = 0; ma < 2; ++ma) {
      const int m0 = mbase + ma * 16;
      #pragma unroll
      for (int v = 0; v < 8; ++v)
        p.out[(size_t)(m0 + v) * p.N + n] = acc[ma][nb][v] + bsum;
    }
  }
}

// ---------------------------------------------------------------------------
// LSTM pointwise: gates [B,4R] (i,f,g,o) + c_in -> h, c  (R = 1024)
// ---------------------------------------------------------------------------
__global__ void lstm_pointwise(const float* __restrict__ g, const float* __restrict__ c_in,
                               float* __restrict__ h_out, float* __restrict__ c_out,
                               float* __restrict__ h_dup) {
  const int idx = blockIdx.x * blockDim.x + threadIdx.x;  // over B*R
  const int b = idx >> 10;
  const int r = idx & 1023;
  const size_t gb = (size_t)b * 4096;
  const float gi = g[gb + r];
  const float gf = g[gb + 1024 + r];
  const float gg = g[gb + 2048 + r];
  const float go = g[gb + 3072 + r];
  const float si = 1.f / (1.f + __expf(-gi));
  const float sf = 1.f / (1.f + __expf(-gf));
  const float so = 1.f / (1.f + __expf(-go));
  const float c2 = sf * c_in[idx] + si * tanhf(gg);
  const float h2 = so * tanhf(c2);
  c_out[idx] = c2;
  h_out[idx] = h2;
  if (h_dup) h_dup[idx] = h2;
}

// ---------------------------------------------------------------------------
// Attention: one block (256 thr) per batch row. H=512, R=1024, L<=64.
// scores_l = tanh(p_feats + att_h) . wa + ba ; softmax ; *mask ; renorm ;
// out = sum_l w_l * feats[:,l,:]
// ---------------------------------------------------------------------------
__global__ __launch_bounds__(256) void attend_kernel(
    const float* __restrict__ att_h, const float* __restrict__ p_feats,
    const float* __restrict__ feats, const float* __restrict__ mask,
    const float* __restrict__ wa, const float* __restrict__ ba,
    float* __restrict__ out, int L) {
  const int b   = blockIdx.x;
  const int tid = threadIdx.x;
  __shared__ float sh_h[512];
  __shared__ float sh_wa[512];
  __shared__ float sc[64];
  __shared__ float sw[64];

  for (int i = tid; i < 512; i += 256) {
    sh_h[i]  = att_h[(size_t)b * 512 + i];
    sh_wa[i] = wa[i];
  }
  __syncthreads();

  const int wave = tid >> 5, lane = tid & 31;
  for (int l = wave; l < L; l += 8) {
    const float* pf = p_feats + ((size_t)b * L + l) * 512;
    float s = 0.f;
    for (int h = lane; h < 512; h += 32)
      s += tanhf(pf[h] + sh_h[h]) * sh_wa[h];
    #pragma unroll
    for (int off = 16; off > 0; off >>= 1) s += __shfl_xor(s, off, 32);
    if (lane == 0) sc[l] = s + ba[0];
  }
  __syncthreads();

  if (tid == 0) {
    float mx = -1e30f;
    for (int l = 0; l < L; ++l) mx = fmaxf(mx, sc[l]);
    float sum = 0.f;
    for (int l = 0; l < L; ++l) { float e = __expf(sc[l] - mx); sw[l] = e; sum += e; }
    const float inv = 1.f / sum;
    float msum = 0.f;
    for (int l = 0; l < L; ++l) {
      float w = sw[l] * inv * mask[(size_t)b * L + l];
      sw[l] = w; msum += w;
    }
    const float inv2 = 1.f / (msum + 1e-8f);
    for (int l = 0; l < L; ++l) sw[l] *= inv2;
  }
  __syncthreads();

  for (int r = tid; r < 1024; r += 256) {
    float acc = 0.f;
    const float* fb = feats + (size_t)b * L * 1024 + r;
    for (int l = 0; l < L; ++l) acc += sw[l] * fb[(size_t)l * 1024];
    out[(size_t)b * 1024 + r] = acc;
  }
}

// ---------------------------------------------------------------------------
// Launch
// ---------------------------------------------------------------------------
extern "C" void kernel_launch(void* const* d_in, const int* in_sizes, int n_in,
                              void* d_out, int out_size, void* d_ws, size_t ws_size,
                              hipStream_t stream) {
  (void)in_sizes; (void)n_in; (void)out_size; (void)ws_size;
  constexpr int Bb = 1024, R = 1024, E = 1024, H = 512;
  constexpr size_t BR = (size_t)Bb * R;

  const float* xt        = (const float*)d_in[0];
  const float* fc_feats  = (const float*)d_in[1];
  const float* state_h   = (const float*)d_in[2];   // [2,B,R]
  const float* state_c   = (const float*)d_in[3];   // [2,B,R]
  const float* feats3[3]  = { (const float*)d_in[4], (const float*)d_in[5], (const float*)d_in[6] };
  const float* pfeats3[3] = { (const float*)d_in[7], (const float*)d_in[8], (const float*)d_in[9] };
  const float* masks3[3]  = { (const float*)d_in[10], (const float*)d_in[11], (const float*)d_in[12] };
  const float* att_Wih  = (const float*)d_in[13];
  const float* att_Whh  = (const float*)d_in[14];
  const float* att_bih  = (const float*)d_in[15];
  const float* att_bhh  = (const float*)d_in[16];
  const float* lang_Wih = (const float*)d_in[17];
  const float* lang_Whh = (const float*)d_in[18];
  const float* lang_bih = (const float*)d_in[19];
  const float* lang_bhh = (const float*)d_in[20];
  const float* Wq3[3] = { (const float*)d_in[21], (const float*)d_in[25], (const float*)d_in[29] };
  const float* bq3[3] = { (const float*)d_in[22], (const float*)d_in[26], (const float*)d_in[30] };
  const float* wa3[3] = { (const float*)d_in[23], (const float*)d_in[27], (const float*)d_in[31] };
  const float* ba3[3] = { (const float*)d_in[24], (const float*)d_in[28], (const float*)d_in[32] };
  const int L3[3] = { 16, 36, 64 };

  // workspace layout (floats): gates[4BR] | att_h x3 [3*B*H] | attended x3 [3*BR]
  float* ws     = (float*)d_ws;
  float* gates  = ws;
  float* atth0  = ws + 4 * BR;
  float* aout0  = atth0 + 3 * (size_t)Bb * H;

  // output layout: output | h_att | h_lang | c_att | c_lang  (each B*R)
  float* out    = (float*)d_out;
  float* h_att  = out + BR;
  float* h_lang = out + 2 * BR;
  float* c_att  = out + 3 * BR;
  float* c_lang = out + 4 * BR;

  // ---- 1) attention-LSTM gates: A = [prev_h | fc | xt | h0], W = [att_Wih | att_Whh]
  GemmParams p1 = {};
  p1.a_ptr[0] = state_h + BR; p1.a_len[0] = R;   // prev_h = state_h[-1]
  p1.a_ptr[1] = fc_feats;     p1.a_len[1] = R;
  p1.a_ptr[2] = xt;           p1.a_len[2] = E;
  p1.a_ptr[3] = state_h;      p1.a_len[3] = R;   // h0
  p1.n_a = 4;
  p1.b_ptr[0] = att_Wih; p1.b_len[0] = E + 2 * R;  // 3072
  p1.b_ptr[1] = att_Whh; p1.b_len[1] = R;
  p1.n_b = 2;
  p1.bias0 = att_bih; p1.bias1 = att_bhh;
  p1.out = gates; p1.M = Bb; p1.N = 4 * R; p1.K = E + 3 * R;  // K = 4096
  gemm_bf16_wmma<<<dim3(p1.N / BN, p1.M / BM), 256, 0, stream>>>(p1);

  lstm_pointwise<<<(int)(BR / 256), 256, 0, stream>>>(gates, state_c, h_att, c_att, nullptr);

  // ---- 2) three attention branches
  for (int br = 0; br < 3; ++br) {
    float* ah = atth0 + (size_t)br * Bb * H;
    float* ao = aout0 + (size_t)br * BR;
    GemmParams pq = {};
    pq.a_ptr[0] = h_att;  pq.a_len[0] = R; pq.n_a = 1;
    pq.b_ptr[0] = Wq3[br]; pq.b_len[0] = R; pq.n_b = 1;
    pq.bias0 = bq3[br]; pq.bias1 = nullptr;
    pq.out = ah; pq.M = Bb; pq.N = H; pq.K = R;
    gemm_bf16_wmma<<<dim3(pq.N / BN, pq.M / BM), 256, 0, stream>>>(pq);
    attend_kernel<<<Bb, 256, 0, stream>>>(ah, pfeats3[br], feats3[br], masks3[br],
                                          wa3[br], ba3[br], ao, L3[br]);
  }

  // ---- 3) language-LSTM gates: A = [h_att | a_attr | a_obj | a_rela | h1]
  GemmParams p2 = {};
  p2.a_ptr[0] = h_att;           p2.a_len[0] = R;
  p2.a_ptr[1] = aout0;           p2.a_len[1] = R;
  p2.a_ptr[2] = aout0 + BR;      p2.a_len[2] = R;
  p2.a_ptr[3] = aout0 + 2 * BR;  p2.a_len[3] = R;
  p2.a_ptr[4] = state_h + BR;    p2.a_len[4] = R;   // h1
  p2.n_a = 5;
  p2.b_ptr[0] = lang_Wih; p2.b_len[0] = 4 * R;
  p2.b_ptr[1] = lang_Whh; p2.b_len[1] = R;
  p2.n_b = 2;
  p2.bias0 = lang_bih; p2.bias1 = lang_bhh;
  p2.out = gates; p2.M = Bb; p2.N = 4 * R; p2.K = 5 * R;  // K = 5120
  gemm_bf16_wmma<<<dim3(p2.N / BN, p2.M / BM), 256, 0, stream>>>(p2);

  lstm_pointwise<<<(int)(BR / 256), 256, 0, stream>>>(gates, state_c + BR,
                                                      h_lang, c_lang, out);
}
